// AttentionBasedPooling_40845138985507
// MI455X (gfx1250) — matrix operand order repaired and checked by reading, not compiled, not run on hardware
//
#include <hip/hip_runtime.h>
#include <hip/hip_bf16.h>

// Problem constants (match setup_inputs; N/S derived from sizes at launch).
#define DD 256   // feature dim
#define HH 128   // hidden dim
#define KT (DD / 32)   // 8 K-tiles of 32
#define NT (HH / 16)   // 8 N-tiles of 16
#define NFRAG (KT * NT)            // 64 B fragments
#define PACKED_BYTES (NFRAG * 32 * 16 * 2)  // 65536 bytes of packed bf16 B
#define PACKED_DW (PACKED_BYTES / 4)        // 16384 dwords

typedef __attribute__((ext_vector_type(16))) __bf16 v16bf;
typedef __attribute__((ext_vector_type(2)))  __bf16 v2bf;
typedef __attribute__((ext_vector_type(8)))  float  v8f;
typedef __attribute__((ext_vector_type(4)))  unsigned u32x4;
typedef __attribute__((ext_vector_type(8)))  int      i32x8;

__device__ __forceinline__ unsigned short f32_to_bf16_bits(float f) {
    union { float f; unsigned u; } c; c.f = f;
    unsigned r = c.u + 0x7FFFu + ((c.u >> 16) & 1u);   // round-to-nearest-even
    return (unsigned short)(r >> 16);
}

// Convert a pair of floats to bf16 using the hardware packed convert if the
// builtin exists; otherwise let the compiler lower the plain cast.
__device__ __forceinline__ void cvt_pair(__bf16* dst, float f0, float f1) {
#if __has_builtin(__builtin_amdgcn_cvt_pk_bf16_f32)
    v2bf p = __builtin_amdgcn_cvt_pk_bf16_f32(f0, f1);
    dst[0] = p[0]; dst[1] = p[1];
#else
    dst[0] = (__bf16)f0; dst[1] = (__bf16)f1;
#endif
}

// Monotone order-preserving float<->uint mapping (for atomicMax on floats).
__device__ __forceinline__ unsigned ford(float f) {
    unsigned u = __float_as_uint(f);
    return (u & 0x80000000u) ? ~u : (u | 0x80000000u);
}
__device__ __forceinline__ float funord(unsigned o) {
    unsigned u = (o & 0x80000000u) ? (o & 0x7FFFFFFFu) : ~o;
    return __uint_as_float(u);
}

// ---------------------------------------------------------------------------
// Kernel 1: zero the context region of d_out, seg_sum, seg_max(ordered).
// ---------------------------------------------------------------------------
__global__ void k_init(float* __restrict__ ctx, float* __restrict__ segsum,
                       unsigned* __restrict__ segmax, int SD, int S) {
    int i = blockIdx.x * blockDim.x + threadIdx.x;
    if (i < SD) ctx[i] = 0.0f;
    if (i < S) { segsum[i] = 0.0f; segmax[i] = 0u; }
}

// ---------------------------------------------------------------------------
// Kernel 2: pack W1 [DD,HH] f32 row-major into bf16 B-operand fragments.
// B 32x16 layout used here (contiguous k, same as A): lane<16 -> col n=lane, K=0..15 ; lane>=16 ->
// col n=lane-16, K=16..31 ; within a lane, element e -> k = hi*16 + e.
// Fragment f = kt*NT + nt, flat: packed[(f*32+lane)*16 + e].
// ---------------------------------------------------------------------------
__global__ void k_pack_w1(const float* __restrict__ W1,
                          unsigned short* __restrict__ packed) {
    int t = blockIdx.x * blockDim.x + threadIdx.x;
    if (t >= NFRAG * 32) return;
    int lane = t & 31, f = t >> 5;
    int kt = f / NT, nt = f % NT;
    int l16 = lane & 15, hi = lane >> 4;
    int n = nt * 16 + l16;
    unsigned short* dst = packed + (size_t)(f * 32 + lane) * 16;
#pragma unroll
    for (int e = 0; e < 16; ++e) {
        int k = kt * 32 + hi * 16 + e;
        dst[e] = f32_to_bf16_bits(W1[(size_t)k * HH + n]);
    }
}

// ---------------------------------------------------------------------------
// Kernel 3: WMMA logits. Block = 256 threads = 8 waves; one 128-row tile per
// block (16 rows per wave) -- single-tile blocks keep the B fragments as
// per-use LDS loads (no LICM -> no VGPR spilling). B is staged ws->LDS with
// one Tensor Data Mover DMA issued by wave 0.
// h = tanh(x@W1 + b1); z = h@W2 + b2  -> zbuf[N]
// ---------------------------------------------------------------------------
__global__ void __launch_bounds__(256)
k_logits(const float* __restrict__ x,
         const unsigned short* __restrict__ packedB,
         const float* __restrict__ b1, const float* __restrict__ W2,
         const float* __restrict__ b2, float* __restrict__ zbuf) {
    __shared__ __align__(16) unsigned char ldsB[PACKED_BYTES];

    // ---- Stage packed B (64KB) ws -> LDS with one TDM tensor_load_to_lds.
    // D# per CDNA5 ISA: 1-D tensor of 16384 dwords,
    // tile == tensor, no multicast, no padding, no iteration.
    if (threadIdx.x < 32) {                       // wave 0 issues the DMA
        unsigned long long ga = (unsigned long long)(uintptr_t)packedB;
        unsigned lds_off = (unsigned)(uintptr_t)&ldsB[0];   // LDS byte offset

        u32x4 g0;
        g0[0] = 1u;                                // count=1, is_restore=0
        g0[1] = lds_off;                           // lds_addr [63:32]
        g0[2] = (unsigned)(ga & 0xFFFFFFFFu);      // global_addr[31:0]
        g0[3] = (unsigned)((ga >> 32) & 0x01FFFFFFu) | (2u << 30); // type=2

        i32x8 g1;
        g1[0] = (int)(2u << 16);                   // data_size=2 (4 bytes)
        g1[1] = (int)((PACKED_DW & 0xFFFFu) << 16);        // dim0 lo16
        g1[2] = (int)((PACKED_DW >> 16) | (1u << 16));     // dim0 hi16, dim1=1 lo
        g1[3] = (int)((PACKED_DW & 0xFFFFu) << 16);        // dim1 hi=0, tile_dim0
        g1[4] = 1;                                 // tile_dim1=1, tile_dim2=0
        g1[5] = (int)PACKED_DW;                    // tensor_dim0_stride lo32
        g1[6] = 0;                                 // stride0 hi16, stride1 lo16
        g1[7] = 0;                                 // stride1 hi32

        asm volatile("tensor_load_to_lds %0, %1"
                     :: "s"(g0), "s"(g1) : "memory");
        __builtin_amdgcn_s_wait_tensorcnt(0);
    }
    __syncthreads();

    const int lane = threadIdx.x & 31;
    const int wave = threadIdx.x >> 5;
    const int l16 = lane & 15;
    const int hi  = lane >> 4;       // 0: rows 0..7 / K-low ; 1: rows 8..15 / K-high
    const int m0  = blockIdx.x * 128 + wave * 16;
    const int row = m0 + l16;                     // A-operand row for this lane
    const float* xr = x + (size_t)row * DD;

    v8f acc[NT];
#pragma unroll
    for (int nt = 0; nt < NT; ++nt) acc[nt] = (v8f){};

#pragma unroll
    for (int kt = 0; kt < KT; ++kt) {
        // A 16x32 bf16 fragment: lane holds row `row`;
        // elements 0..15 -> k = kt*32 + hi*16 + e.
        const int k0 = kt * 32 + hi * 16;
        __bf16 tmp[16];
#pragma unroll
        for (int e = 0; e < 8; e += 2) {
            cvt_pair(&tmp[e],     xr[k0 + e],      xr[k0 + e + 1]);
            cvt_pair(&tmp[e + 8], xr[k0 + 8 + e], xr[k0 + 8 + e + 1]);
        }
        v16bf a;
        __builtin_memcpy(&a, tmp, 32);
#pragma unroll
        for (int nt = 0; nt < NT; ++nt) {
            const v16bf b =
                *(const v16bf*)(ldsB + ((kt * NT + nt) * 32 + lane) * 32);
            acc[nt] = __builtin_amdgcn_wmma_f32_16x16x32_bf16(
                false, a, false, b, (short)0, acc[nt], false, false);
        }
    }

    // Epilogue: p[j] = sum_n tanh(h[m,n] + b1[n]) * W2[n]
    // C layout: lane l16 = col, vgpr j = row within the `hi` half;
    // butterfly-reduce over the 16-lane half (masks <= 8 stay in-half).
    float p[8];
#pragma unroll
    for (int j = 0; j < 8; ++j) p[j] = 0.0f;
#pragma unroll
    for (int nt = 0; nt < NT; ++nt) {
        const int n = nt * 16 + l16;
        const float bn = b1[n];
        const float wn = W2[n];
#pragma unroll
        for (int j = 0; j < 8; ++j)
            p[j] += tanhf(acc[nt][j] + bn) * wn;
    }
#pragma unroll
    for (int mask = 1; mask <= 8; mask <<= 1)
#pragma unroll
        for (int j = 0; j < 8; ++j)
            p[j] += __shfl_xor(p[j], mask, 32);

    if (l16 == 0) {
        const float b2v = b2[0];
#pragma unroll
        for (int j = 0; j < 8; ++j)
            zbuf[m0 + hi * 8 + j] = p[j] + b2v;
    }
}

// ---------------------------------------------------------------------------
// Kernel 4: segment max via order-preserving uint atomicMax.
// ---------------------------------------------------------------------------
__global__ void k_segmax(const float* __restrict__ z,
                         const int* __restrict__ sid,
                         unsigned* __restrict__ segmax, int N) {
    int i = blockIdx.x * blockDim.x + threadIdx.x;
    if (i >= N) return;
    atomicMax(&segmax[sid[i]], ford(z[i]));
}

// ---------------------------------------------------------------------------
// Kernel 5: e = exp(z - segmax); atomic add into seg_sum.
// ---------------------------------------------------------------------------
__global__ void k_expsum(const float* __restrict__ z,
                         const int* __restrict__ sid,
                         const unsigned* __restrict__ segmax,
                         float* __restrict__ ebuf,
                         float* __restrict__ segsum, int N) {
    int i = blockIdx.x * blockDim.x + threadIdx.x;
    if (i >= N) return;
    int s = sid[i];
    float ev = expf(z[i] - funord(segmax[s]));
    ebuf[i] = ev;
    atomicAdd(&segsum[s], ev);
}

// ---------------------------------------------------------------------------
// Kernel 6: w = e/segsum -> attn out; per-block run-accumulated weighted sum
// of x into context via one float atomic per segment boundary (sids sorted).
// blockDim.x == DD; each thread owns one feature dim; block covers 256 rows.
// ---------------------------------------------------------------------------
__global__ void __launch_bounds__(DD)
k_finalize(const float* __restrict__ x, const int* __restrict__ sid,
           const float* __restrict__ ebuf, const float* __restrict__ segsum,
           float* __restrict__ ctx, float* __restrict__ attn) {
    __shared__ float wsh[256];
    __shared__ int   ssh[256];
    const int tid = threadIdx.x;
    const int r0  = blockIdx.x * 256;

    {
        int row = r0 + tid;
        int s = sid[row];
        float w = ebuf[row] / segsum[s];
        attn[row] = w;
        wsh[tid] = w;
        ssh[tid] = s;
    }
    __syncthreads();

    float acc = 0.0f;
    int prev = ssh[0];
#pragma unroll 4
    for (int i = 0; i < 256; ++i) {
        int si = ssh[i];
        if (si != prev) {
            atomicAdd(&ctx[(size_t)prev * DD + tid], acc);
            acc = 0.0f;
            prev = si;
        }
        acc += wsh[i] * x[(size_t)(r0 + i) * DD + tid];
    }
    atomicAdd(&ctx[(size_t)prev * DD + tid], acc);
}

// ---------------------------------------------------------------------------
extern "C" void kernel_launch(void* const* d_in, const int* in_sizes, int n_in,
                              void* d_out, int out_size, void* d_ws, size_t ws_size,
                              hipStream_t stream) {
    const float* x   = (const float*)d_in[0];
    const int*   sid = (const int*)d_in[1];
    // d_in[2] = num_segments (device scalar; S derived from sizes instead)
    const float* W1  = (const float*)d_in[3];
    const float* b1  = (const float*)d_in[4];
    const float* W2  = (const float*)d_in[5];
    const float* b2  = (const float*)d_in[6];

    const int N = in_sizes[1];               // 524288
    const int S = (out_size - N) / DD;       // 2048

    float* ctx  = (float*)d_out;             // [S, DD]
    float* attn = ctx + (size_t)S * DD;      // [N]

    // Workspace layout
    char* w = (char*)d_ws;
    unsigned short* packedB = (unsigned short*)w;            // 64 KB
    float* zbuf   = (float*)(w + PACKED_BYTES);              // N floats
    float* ebuf   = zbuf + N;                                // N floats
    float* segsum = ebuf + N;                                // S floats
    unsigned* segmax = (unsigned*)(segsum + S);              // S uints
    (void)ws_size; (void)n_in;

    // 1) init
    {
        int SD = S * DD;
        int blocks = (SD + 255) / 256;
        k_init<<<blocks, 256, 0, stream>>>(ctx, segsum, segmax, SD, S);
    }
    // 2) pack W1 -> bf16 B fragments
    {
        int threads = NFRAG * 32;            // 2048
        k_pack_w1<<<(threads + 255) / 256, 256, 0, stream>>>(W1, packedB);
    }
    // 3) WMMA logits (128 rows / block; TDM-staged B in LDS)
    k_logits<<<N / 128, 256, 0, stream>>>(x, packedB, b1, W2, b2, zbuf);
    // 4) segment max
    k_segmax<<<(N + 255) / 256, 256, 0, stream>>>(zbuf, sid, segmax, N);
    // 5) exp + segment sum
    k_expsum<<<(N + 255) / 256, 256, 0, stream>>>(zbuf, sid, segmax, ebuf, segsum, N);
    // 6) attention weights + context accumulation
    k_finalize<<<N / 256, DD, 0, stream>>>(x, sid, ebuf, segsum, ctx, attn);
}
